// flashUNet_74345883894471
// MI455X (gfx1250) — compile-verified, matrix operand order checked by reading, not run on hardware
//
#include <hip/hip_runtime.h>
#include <hip/hip_bf16.h>

// ---------------------------------------------------------------------------
// Gated-attention edge block for MI455X (gfx1250, wave32, WMMA).
//   B=4, IC=512, H=W=128, INTER=16, C3=48.
// Heavy K=512 channel reductions run on v_wmma_f32_16x16x32_bf16 (fp32 acc).
// dil/ero/ex staged NHWC bf16; 18 weight matrices resident in 288KB LDS.
// 8 waves per workgroup (one block per image row) to maximize waves/WGP
// under the 288KB-LDS-per-WGP constraint.
// ---------------------------------------------------------------------------

#define BN_   4
#define ICN   512
#define HN    128
#define WN    128
#define HWN   16384
#define INTERN 16

typedef __attribute__((ext_vector_type(16))) __bf16 v16bf;
typedef __attribute__((ext_vector_type(8)))  float  v8f;

__device__ __forceinline__ unsigned short f2bf(float f) {
  unsigned u = __float_as_uint(f);
  u = u + 0x7FFFu + ((u >> 16) & 1u);   // round-to-nearest-even
  return (unsigned short)(u >> 16);
}

// Pack one 16x32 bf16 fragment (A: rows=pixels, B: rows=oc; both stored
// row-major with row stride 512).  ISA layout: lane half h, VGPR r holds the
// K-pair starting at ((r&4)?16:0) + 8*h + 2*(r&3).
__device__ __forceinline__ v16bf load_frag(const unsigned short* row0,
                                           int kbase, int lane) {
  const unsigned short* p = row0 + (size_t)(lane & 15) * ICN + kbase;
  int h = lane >> 4;
  union { v16bf v; unsigned int u[8]; } f;
#pragma unroll
  for (int r = 0; r < 8; ++r) {
    int kk = ((r & 4) ? 16 : 0) + h * 8 + (r & 3) * 2;
    f.u[r] = *(const unsigned int*)(p + kk);
  }
  return f.v;
}

__device__ __forceinline__ v8f wmma_bf16(v16bf a, v16bf b, v8f c) {
  return __builtin_amdgcn_wmma_f32_16x16x32_bf16(false, a, false, b,
                                                 (short)0, c, false, false);
}

// ---------------------------------------------------------------------------
// prep: repack weights to bf16 [oc][ic] per tap; fuse BN scale/shift.
// Wd/We: mat 0 = 1x1, mats 1..3 = 3x1 taps, mats 4..8 = 5x1 taps.
// ---------------------------------------------------------------------------
__global__ void prep_kernel(const float* Dw1, const float* Dw31, const float* Dw51,
                            const float* Ew1, const float* Ew31, const float* Ew51,
                            const float* qw, const float* kw, const float* vw,
                            const float* Dg, const float* Dbe, const float* Dm, const float* Dv,
                            const float* Eg, const float* Ebe, const float* Em, const float* Ev,
                            unsigned short* Wd, unsigned short* We,
                            unsigned short* Wqkv, float* bn) {
  int tid = blockIdx.x * blockDim.x + threadIdx.x;
  int nt = gridDim.x * blockDim.x;
  for (int i = tid; i < 9 * 16 * 512; i += nt) {
    int ic = i & 511, oc = (i >> 9) & 15, mat = i >> 13;
    float vd, ve;
    if (mat == 0)      { vd = Dw1[oc * 512 + ic];                 ve = Ew1[oc * 512 + ic]; }
    else if (mat < 4)  { int t = mat - 1;
                         vd = Dw31[(oc * 512 + ic) * 3 + t];      ve = Ew31[(oc * 512 + ic) * 3 + t]; }
    else               { int t = mat - 4;
                         vd = Dw51[(oc * 512 + ic) * 5 + t];      ve = Ew51[(oc * 512 + ic) * 5 + t]; }
    Wd[i] = f2bf(vd);  We[i] = f2bf(ve);
  }
  for (int i = tid; i < 3 * 16 * 512; i += nt) {
    int ic = i & 511, oc = (i >> 9) & 15, m = i >> 13;
    const float* w = (m == 0) ? qw : ((m == 1) ? kw : vw);
    Wqkv[i] = f2bf(w[oc * 512 + ic]);
  }
  for (int i = tid; i < 48; i += nt) {
    float sd = Dg[i] * rsqrtf(Dv[i] + 1e-5f);
    bn[i] = sd;        bn[48 + i]  = Dbe[i] - Dm[i] * sd;
    float se = Eg[i] * rsqrtf(Ev[i] + 1e-5f);
    bn[96 + i] = se;   bn[144 + i] = Ebe[i] - Em[i] * se;
  }
}

// ---------------------------------------------------------------------------
// NCHW f32 -> NHWC bf16 with fused 3x3 dilate (max) / erode (min).
// LDS tile transpose so both global reads and writes are coalesced.
// grid: B*H*(W/64) = 1024 blocks, 256 threads.
// ---------------------------------------------------------------------------
__global__ void dil_ero_nhwc(const float* __restrict__ x,
                             unsigned short* __restrict__ dil,
                             unsigned short* __restrict__ ero) {
  __shared__ unsigned short t0[32][72];
  __shared__ unsigned short t1[32][72];
  int bid = blockIdx.x;
  int xseg = bid & 1, y = (bid >> 1) & 127, b = bid >> 8;
  int x0 = xseg * 64;
  int t = threadIdx.x;
  for (int c0 = 0; c0 < 512; c0 += 32) {
    int lx = t & 63, c4 = t >> 6;
#pragma unroll
    for (int pp = 0; pp < 8; ++pp) {
      int cl = pp * 4 + c4, c = c0 + cl, xx0 = x0 + lx;
      float mn = 3.4e38f, mx = -3.4e38f;
#pragma unroll
      for (int dy = -1; dy <= 1; ++dy) {
        int yy = y + dy; if (yy < 0 || yy > 127) continue;
#pragma unroll
        for (int dxx = -1; dxx <= 1; ++dxx) {
          int xx = xx0 + dxx; if (xx < 0 || xx > 127) continue;
          float v = x[(((size_t)b * 512 + c) * 128 + yy) * 128 + xx];
          mn = fminf(mn, v); mx = fmaxf(mx, v);
        }
      }
      t0[cl][lx] = f2bf(mx);
      t1[cl][lx] = f2bf(mn);
    }
    __syncthreads();
    int cl = t & 31, p4 = t >> 5;
#pragma unroll
    for (int pp = 0; pp < 8; ++pp) {
      int p = pp * 8 + p4;
      size_t o = (((size_t)b * 128 + y) * 128 + x0 + p) * 512 + c0 + cl;
      dil[o] = t0[cl][p];
      ero[o] = t1[cl][p];
    }
    __syncthreads();
  }
}

// Plain NCHW f32 -> NHWC bf16 transpose (for ex).
__global__ void nhwc_bf16(const float* __restrict__ x,
                          unsigned short* __restrict__ dst) {
  __shared__ unsigned short t0[32][72];
  int bid = blockIdx.x;
  int xseg = bid & 1, y = (bid >> 1) & 127, b = bid >> 8;
  int x0 = xseg * 64;
  int t = threadIdx.x;
  for (int c0 = 0; c0 < 512; c0 += 32) {
    int lx = t & 63, c4 = t >> 6;
#pragma unroll
    for (int pp = 0; pp < 8; ++pp) {
      int cl = pp * 4 + c4, c = c0 + cl;
      t0[cl][lx] = f2bf(x[(((size_t)b * 512 + c) * 128 + y) * 128 + x0 + lx]);
    }
    __syncthreads();
    int cl = t & 31, p4 = t >> 5;
#pragma unroll
    for (int pp = 0; pp < 8; ++pp) {
      int p = pp * 8 + p4;
      dst[(((size_t)b * 128 + y) * 128 + x0 + p) * 512 + c0 + cl] = t0[cl][p];
    }
    __syncthreads();
  }
}

// ---------------------------------------------------------------------------
// The heavy kernel: all six 512->16 strip-conv GEMMs via WMMA.
// One wave32 per 16-pixel tile, 8 waves per block = one image row per block.
// 18 weight matrices (288KB) staged in LDS -> 8 resident waves per WGP.
// Vertical taps +-2 rows, zero-pad via wave-uniform branch skip.
// Also fuses `difference` out of the six accumulators.
// grid: B*H = 512 blocks, 256 threads, 294912B dynamic LDS.
// ---------------------------------------------------------------------------
__global__ void striped_wmma(const unsigned short* __restrict__ dil,
                             const unsigned short* __restrict__ ero,
                             const unsigned short* __restrict__ wg,   // Wd||We
                             const float* Db1, const float* Db31, const float* Db51,
                             const float* Eb1, const float* Eb31, const float* Eb51,
                             float* __restrict__ Dx1o, float* __restrict__ Dx31o,
                             float* __restrict__ Dx51o,
                             float* __restrict__ Ex1o, float* __restrict__ Ex31o,
                             float* __restrict__ Ex51o,
                             float* __restrict__ diffo) {
  extern __shared__ unsigned short sw[];   // 18 * 16 * 512 bf16 = 294912 B
  {
    uint4* d = (uint4*)sw;
    const uint4* s = (const uint4*)wg;
    for (int i = threadIdx.x; i < 18432; i += blockDim.x) d[i] = s[i];
  }
  __syncthreads();

  int bid = blockIdx.x;                 // b*128 + y
  int y = bid & 127, b = bid >> 7;
  int wave = threadIdx.x >> 5, lane = threadIdx.x & 31;
  int x0 = wave * 16;

  v8f a0 = {0,0,0,0,0,0,0,0}, a1 = a0, a2 = a0, a3 = a0, a4 = a0, a5 = a0;

  for (int kk = 0; kk < 512; kk += 32) {
#pragma unroll
    for (int dy = -2; dy <= 2; ++dy) {
      int yy = y + dy;
      if (yy < 0 || yy >= 128) continue;          // zero-pad tap (wave-uniform)
      size_t pb = (((size_t)b * 128 + yy) * 128 + x0) * 512;
      v16bf ad = load_frag(dil + pb, kk, lane);
      v16bf ae = load_frag(ero + pb, kk, lane);
      a2 = wmma_bf16(ad, load_frag(sw + (size_t)(4 + dy + 2) * 8192, kk, lane), a2);
      a5 = wmma_bf16(ae, load_frag(sw + (size_t)(13 + dy + 2) * 8192, kk, lane), a5);
      if (dy >= -1 && dy <= 1) {
        a1 = wmma_bf16(ad, load_frag(sw + (size_t)(1 + dy + 1) * 8192, kk, lane), a1);
        a4 = wmma_bf16(ae, load_frag(sw + (size_t)(10 + dy + 1) * 8192, kk, lane), a4);
        if (dy == 0) {
          a0 = wmma_bf16(ad, load_frag(sw, kk, lane), a0);
          a3 = wmma_bf16(ae, load_frag(sw + (size_t)9 * 8192, kk, lane), a3);
        }
      }
    }
  }

  int n = lane & 15, h = lane >> 4;
  int q0 = y * 128 + x0;
  float b1 = Db1[n], b31 = Db31[n], b51 = Db51[n];
  float c1 = Eb1[n], c31 = Eb31[n], c51 = Eb51[n];
  size_t cb = ((size_t)b * 16 + n) * HWN;
#pragma unroll
  for (int r = 0; r < 8; ++r) {
    size_t q = cb + q0 + r + 8 * h;
    float d0 = a0[r] + b1, d1 = a1[r] + b31, d2 = a2[r] + b51;
    float e0 = a3[r] + c1, e1 = a4[r] + c31, e2 = a5[r] + c51;
    Dx1o[q] = d0; Dx31o[q] = d1; Dx51o[q] = d2;
    Ex1o[q] = e0; Ex31o[q] = e1; Ex51o[q] = e2;
    diffo[q] = 2.f * d0 + d1 + d2 - e1 - e2 - 2.f * e0;   // difference, bias incl.
  }
}

// ---------------------------------------------------------------------------
// s = relu(bn(concat[x1, x3113, x5115]))_D - same_E   (48 channels)
// 1x3 / 1x5 horizontal convs on the 16-ch strip outputs.
// ---------------------------------------------------------------------------
__global__ void s_kernel(const float* __restrict__ Dx1, const float* __restrict__ Dx31,
                         const float* __restrict__ Dx51,
                         const float* __restrict__ Ex1, const float* __restrict__ Ex31,
                         const float* __restrict__ Ex51,
                         const float* Dw13, const float* Db13,
                         const float* Dw15, const float* Db15,
                         const float* Ew13, const float* Eb13,
                         const float* Ew15, const float* Eb15,
                         const float* __restrict__ bn, float* __restrict__ s) {
  int id = blockIdx.x * blockDim.x + threadIdx.x;   // 65536
  int b = id >> 14, q = id & 16383, x = q & 127;
  size_t ib = (size_t)b * 16 * HWN;
  size_t sb = (size_t)b * 48 * HWN;

#pragma unroll
  for (int oc = 0; oc < 16; ++oc) {
    float fd = fmaxf(Dx1[ib + oc * HWN + q] * bn[oc] + bn[48 + oc], 0.f);
    float fe = fmaxf(Ex1[ib + oc * HWN + q] * bn[96 + oc] + bn[144 + oc], 0.f);
    s[sb + (size_t)oc * HWN + q] = fd - fe;
  }
  {
    float ad[16], ae[16];
#pragma unroll
    for (int oc = 0; oc < 16; ++oc) { ad[oc] = Db13[oc]; ae[oc] = Eb13[oc]; }
    for (int t = -1; t <= 1; ++t) {
      int xx = x + t; if (xx < 0 || xx > 127) continue;
      int qq = q + t;
      for (int ic = 0; ic < 16; ++ic) {
        float xd = Dx31[ib + ic * HWN + qq];
        float xe = Ex31[ib + ic * HWN + qq];
#pragma unroll
        for (int oc = 0; oc < 16; ++oc) {
          ad[oc] += Dw13[(oc * 16 + ic) * 3 + t + 1] * xd;
          ae[oc] += Ew13[(oc * 16 + ic) * 3 + t + 1] * xe;
        }
      }
    }
#pragma unroll
    for (int oc = 0; oc < 16; ++oc) {
      int c = 16 + oc;
      float fd = fmaxf(ad[oc] * bn[c] + bn[48 + c], 0.f);
      float fe = fmaxf(ae[oc] * bn[96 + c] + bn[144 + c], 0.f);
      s[sb + (size_t)c * HWN + q] = fd - fe;
    }
  }
  {
    float ad[16], ae[16];
#pragma unroll
    for (int oc = 0; oc < 16; ++oc) { ad[oc] = Db15[oc]; ae[oc] = Eb15[oc]; }
    for (int t = -2; t <= 2; ++t) {
      int xx = x + t; if (xx < 0 || xx > 127) continue;
      int qq = q + t;
      for (int ic = 0; ic < 16; ++ic) {
        float xd = Dx51[ib + ic * HWN + qq];
        float xe = Ex51[ib + ic * HWN + qq];
#pragma unroll
        for (int oc = 0; oc < 16; ++oc) {
          ad[oc] += Dw15[(oc * 16 + ic) * 5 + t + 2] * xd;
          ae[oc] += Ew15[(oc * 16 + ic) * 5 + t + 2] * xe;
        }
      }
    }
#pragma unroll
    for (int oc = 0; oc < 16; ++oc) {
      int c = 32 + oc;
      float fd = fmaxf(ad[oc] * bn[c] + bn[48 + c], 0.f);
      float fe = fmaxf(ae[oc] * bn[96 + c] + bn[144 + c], 0.f);
      s[sb + (size_t)c * HWN + q] = fd - fe;
    }
  }
}

// ---------------------------------------------------------------------------
// edge path: depthwise 3x3 (48ch) -> pointwise 48->16 -> relu ->
// per-pixel mean/max over 16 -> Wedge = sigmoid(cw0*mean + cw1*max + cb)
// ---------------------------------------------------------------------------
__global__ void edge_kernel(const float* __restrict__ s,
                            const float* dww, const float* dwb,
                            const float* pww, const float* pwb,
                            const float* cw, const float* cb,
                            float* __restrict__ wedge) {
  int id = blockIdx.x * blockDim.x + threadIdx.x;   // 65536
  int b = id >> 14, q = id & 16383, y = q >> 7, x = q & 127;
  size_t sb = (size_t)b * 48 * HWN;
  float e[16];
#pragma unroll
  for (int oc = 0; oc < 16; ++oc) e[oc] = pwb[oc];
  for (int c = 0; c < 48; ++c) {
    float acc = dwb[c];
#pragma unroll
    for (int dy = -1; dy <= 1; ++dy) {
      int yy = y + dy; if (yy < 0 || yy > 127) continue;
#pragma unroll
      for (int dxx = -1; dxx <= 1; ++dxx) {
        int xx = x + dxx; if (xx < 0 || xx > 127) continue;
        acc += dww[(c * 3 + dy + 1) * 3 + dxx + 1] * s[sb + (size_t)c * HWN + yy * 128 + xx];
      }
    }
#pragma unroll
    for (int oc = 0; oc < 16; ++oc) e[oc] += pww[oc * 48 + c] * acc;
  }
  float mean = 0.f, mx = -3.4e38f;
#pragma unroll
  for (int oc = 0; oc < 16; ++oc) {
    float v = fmaxf(e[oc], 0.f);
    mean += v; mx = fmaxf(mx, v);
  }
  mean *= (1.f / 16.f);
  float z = cw[0] * mean + cw[1] * mx + cb[0];
  wedge[(size_t)b * HWN + q] = 1.f / (1.f + __expf(-z));
}

// ---------------------------------------------------------------------------
// Q/K/V 512->16 GEMMs via WMMA; V gated by (1 + Wedge) at store.
// grid: B*H = 512 blocks, 256 threads (one wave per 16-pixel tile).
// ---------------------------------------------------------------------------
__global__ void qkv_wmma(const unsigned short* __restrict__ exT,
                         const unsigned short* __restrict__ Wqkv,
                         const float* qb, const float* kb, const float* vb,
                         const float* __restrict__ wedge,
                         float* __restrict__ Qo, float* __restrict__ Ko,
                         float* __restrict__ Vo) {
  int bid = blockIdx.x;                 // b*128 + y
  int y = bid & 127, b = bid >> 7;
  int wave = threadIdx.x >> 5, lane = threadIdx.x & 31;
  int x0 = wave * 16;
  v8f aq = {0,0,0,0,0,0,0,0}, ak = aq, av = aq;
  size_t pb = (((size_t)b * 128 + y) * 128 + x0) * 512;
  for (int kk = 0; kk < 512; kk += 32) {
    v16bf a = load_frag(exT + pb, kk, lane);
    aq = wmma_bf16(a, load_frag(Wqkv,         kk, lane), aq);
    ak = wmma_bf16(a, load_frag(Wqkv + 8192,  kk, lane), ak);
    av = wmma_bf16(a, load_frag(Wqkv + 16384, kk, lane), av);
  }
  int n = lane & 15, h = lane >> 4;
  int q0 = y * 128 + x0;
  float bq = qb[n], bk = kb[n], bv = vb[n];
  size_t cb = ((size_t)b * 16 + n) * HWN;
  size_t wb = (size_t)b * HWN;
#pragma unroll
  for (int r = 0; r < 8; ++r) {
    int q = q0 + r + 8 * h;
    Qo[cb + q] = aq[r] + bq;
    Ko[cb + q] = ak[r] + bk;
    Vo[cb + q] = (av[r] + bv) * (1.f + wedge[wb + q]);
  }
}

// ---------------------------------------------------------------------------
// M[b,i,j] = sum_p Q[b,i,p] * (Kflat[b,p*16+j] + Dflat[b,p*16+j])
// (weight + gate, torch-.view faithful). One block per (b,i,j).
// ---------------------------------------------------------------------------
__global__ void mm_kernel(const float* __restrict__ Qc, const float* __restrict__ Kc,
                          const float* __restrict__ Df, float* __restrict__ M) {
  __shared__ float red[256];
  int bid = blockIdx.x;              // 4*16*16 = 1024
  int b = bid >> 8, ij = bid & 255, i = ij >> 4, j = ij & 15;
  const float* Qp = Qc + ((size_t)b * 16 + i) * HWN;
  const float* Kp = Kc + (size_t)b * 16 * HWN;
  const float* Dp = Df + (size_t)b * 16 * HWN;
  float part = 0.f;
  for (int p = threadIdx.x; p < HWN; p += 256) {
    float qv = Qp[p];
    part += qv * (Kp[p * 16 + j] + Dp[p * 16 + j]);
  }
  red[threadIdx.x] = part;
  __syncthreads();
  for (int st = 128; st > 0; st >>= 1) {
    if (threadIdx.x < st) red[threadIdx.x] += red[threadIdx.x + st];
    __syncthreads();
  }
  if (threadIdx.x == 0) M[bid] = red[0];
}

// softmax over i (dim=1) per (b, j)
__global__ void softmax_kernel(const float* __restrict__ M, float* __restrict__ Wt) {
  int t = threadIdx.x;
  if (t >= 64) return;
  int b = t >> 4, j = t & 15;
  float v[16], mx = -3.4e38f;
#pragma unroll
  for (int i = 0; i < 16; ++i) { v[i] = M[b * 256 + i * 16 + j]; mx = fmaxf(mx, v[i]); }
  float sum = 0.f;
#pragma unroll
  for (int i = 0; i < 16; ++i) { v[i] = __expf(v[i] - mx); sum += v[i]; }
  float inv = 1.f / sum;
#pragma unroll
  for (int i = 0; i < 16; ++i) Wt[b * 256 + i * 16 + j] = v[i] * inv;
}

// atten[b,i,p] = sum_j Wt[b,i,j] * V[b,j,p]
__global__ void atten_kernel(const float* __restrict__ Wt, const float* __restrict__ Vc,
                             float* __restrict__ att) {
  int id = blockIdx.x * 256 + threadIdx.x;    // 4*16*16384
  int b = id >> 18, rem = id & 262143, i = rem >> 14, p = rem & 16383;
  const float* Vp = Vc + (size_t)b * 16 * HWN;
  float a = 0.f;
#pragma unroll
  for (int j = 0; j < 16; ++j) a += Wt[b * 256 + i * 16 + j] * Vp[(size_t)j * HWN + p];
  att[((size_t)b * 16 + i) * HWN + p] = a;
}

// out[b,oc,q] = rb[oc] + dx[b,oc,q] + sum_c rw[oc,c] * attenX[b,c,q]
// attenX[b,c,q] = atten[b, q%16, c*1024 + q/16]   (transpose+view faithful)
__global__ void out_kernel(const float* __restrict__ att, const float* __restrict__ rw,
                           const float* __restrict__ rb, const float* __restrict__ dx,
                           float* __restrict__ out) {
  int id = blockIdx.x * 256 + threadIdx.x;    // 65536
  int b = id >> 14, q = id & 16383;
  float ax[16];
  const float* ab = att + (size_t)b * 16 * HWN + (size_t)(q & 15) * HWN + (q >> 4);
#pragma unroll
  for (int c = 0; c < 16; ++c) ax[c] = ab[(size_t)c * 1024];
  size_t base = ((size_t)b * 512) * HWN + q;
  for (int oc = 0; oc < 512; ++oc) {
    float acc = rb[oc];
#pragma unroll
    for (int c = 0; c < 16; ++c) acc += rw[oc * 16 + c] * ax[c];
    out[base + (size_t)oc * HWN] = acc + dx[base + (size_t)oc * HWN];
  }
}

// ---------------------------------------------------------------------------
extern "C" void kernel_launch(void* const* d_in, const int* in_sizes, int n_in,
                              void* d_out, int out_size, void* d_ws, size_t ws_size,
                              hipStream_t stream) {
  (void)in_sizes; (void)n_in; (void)out_size; (void)ws_size;
  const float* ex   = (const float*)d_in[0];
  const float* dx   = (const float*)d_in[1];
  const float* Dw1  = (const float*)d_in[2];
  const float* Db1  = (const float*)d_in[3];
  const float* Dw31 = (const float*)d_in[4];
  const float* Db31 = (const float*)d_in[5];
  const float* Dw13 = (const float*)d_in[6];
  const float* Db13 = (const float*)d_in[7];
  const float* Dw51 = (const float*)d_in[8];
  const float* Db51 = (const float*)d_in[9];
  const float* Dw15 = (const float*)d_in[10];
  const float* Db15 = (const float*)d_in[11];
  const float* Dg   = (const float*)d_in[12];
  const float* Dbe  = (const float*)d_in[13];
  const float* Dm   = (const float*)d_in[14];
  const float* Dv   = (const float*)d_in[15];
  const float* Ew1  = (const float*)d_in[16];
  const float* Eb1  = (const float*)d_in[17];
  const float* Ew31 = (const float*)d_in[18];
  const float* Eb31 = (const float*)d_in[19];
  const float* Ew13 = (const float*)d_in[20];
  const float* Eb13 = (const float*)d_in[21];
  const float* Ew51 = (const float*)d_in[22];
  const float* Eb51 = (const float*)d_in[23];
  const float* Ew15 = (const float*)d_in[24];
  const float* Eb15 = (const float*)d_in[25];
  const float* Eg   = (const float*)d_in[26];
  const float* Ebe  = (const float*)d_in[27];
  const float* Em   = (const float*)d_in[28];
  const float* Ev   = (const float*)d_in[29];
  const float* dww  = (const float*)d_in[30];
  const float* dwb  = (const float*)d_in[31];
  const float* pww  = (const float*)d_in[32];
  const float* pwb  = (const float*)d_in[33];
  const float* qw   = (const float*)d_in[34];
  const float* qb   = (const float*)d_in[35];
  const float* kw   = (const float*)d_in[36];
  const float* kb   = (const float*)d_in[37];
  const float* vw   = (const float*)d_in[38];
  const float* vb   = (const float*)d_in[39];
  const float* rw   = (const float*)d_in[40];
  const float* rb   = (const float*)d_in[41];
  const float* cw   = (const float*)d_in[42];
  const float* cb   = (const float*)d_in[43];

  char* ws = (char*)d_ws;
  const size_t SZ_NHWC = (size_t)BN_ * HN * WN * ICN * 2;   // 67108864 B
  unsigned short* dil  = (unsigned short*)(ws);
  unsigned short* ero  = (unsigned short*)(ws + SZ_NHWC);
  unsigned short* exT  = (unsigned short*)(ws + 2 * SZ_NHWC);
  unsigned short* Wd   = (unsigned short*)(ws + 3 * SZ_NHWC);
  unsigned short* We   = Wd + (size_t)9 * 16 * 512;          // contiguous after Wd
  unsigned short* Wqkv = We + (size_t)9 * 16 * 512;
  float* bn   = (float*)(Wqkv + (size_t)3 * 16 * 512);
  float* Dx1  = bn + 256;
  float* Dx31 = Dx1  + (size_t)BN_ * 16 * HWN;
  float* Dx51 = Dx31 + (size_t)BN_ * 16 * HWN;
  float* Ex1  = Dx51 + (size_t)BN_ * 16 * HWN;
  float* Ex31 = Ex1  + (size_t)BN_ * 16 * HWN;
  float* Ex51 = Ex31 + (size_t)BN_ * 16 * HWN;
  float* diff = Ex51 + (size_t)BN_ * 16 * HWN;
  float* Qc   = diff + (size_t)BN_ * 16 * HWN;
  float* Kc   = Qc   + (size_t)BN_ * 16 * HWN;
  float* Vc   = Kc   + (size_t)BN_ * 16 * HWN;
  float* att  = Vc   + (size_t)BN_ * 16 * HWN;
  float* sbuf = att  + (size_t)BN_ * 16 * HWN;
  float* wedge = sbuf + (size_t)BN_ * 48 * HWN;
  float* Mb   = wedge + (size_t)BN_ * HWN;
  float* Wt   = Mb + 1024;

  prep_kernel<<<64, 256, 0, stream>>>(Dw1, Dw31, Dw51, Ew1, Ew31, Ew51,
                                      qw, kw, vw, Dg, Dbe, Dm, Dv,
                                      Eg, Ebe, Em, Ev, Wd, We, Wqkv, bn);
  dil_ero_nhwc<<<1024, 256, 0, stream>>>(dx, dil, ero);
  nhwc_bf16<<<1024, 256, 0, stream>>>(ex, exT);
  striped_wmma<<<512, 256, 294912, stream>>>(dil, ero, Wd,
                                             Db1, Db31, Db51, Eb1, Eb31, Eb51,
                                             Dx1, Dx31, Dx51, Ex1, Ex31, Ex51, diff);
  s_kernel<<<256, 256, 0, stream>>>(Dx1, Dx31, Dx51, Ex1, Ex31, Ex51,
                                    Dw13, Db13, Dw15, Db15,
                                    Ew13, Eb13, Ew15, Eb15, bn, sbuf);
  edge_kernel<<<256, 256, 0, stream>>>(sbuf, dww, dwb, pww, pwb, cw, cb, wedge);
  qkv_wmma<<<512, 256, 0, stream>>>(exT, Wqkv, qb, kb, vb, wedge, Qc, Kc, Vc);
  mm_kernel<<<1024, 256, 0, stream>>>(Qc, Kc, diff, Mb);
  softmax_kernel<<<1, 64, 0, stream>>>(Mb, Wt);
  atten_kernel<<<4096, 256, 0, stream>>>(Wt, Vc, att);
  out_kernel<<<256, 256, 0, stream>>>(att, rw, rb, dx, (float*)d_out);
}
